// ConvVAE2d_10531259809831
// MI455X (gfx1250) — matrix operand order repaired and checked by reading, not compiled, not run on hardware
//
#include <hip/hip_runtime.h>
#include <hip/hip_bf16.h>

typedef __bf16 bf16;
typedef __attribute__((ext_vector_type(16))) __bf16 v16bf;
typedef __attribute__((ext_vector_type(8)))  float  v8f;

// Problem dims
#define NCLS 10
#define PDIM 768          // CH*K*K
#define MT   1024         // M tiles of 16 (M = 16384, 16 rows per batch elem)
#define KT1  25           // K=778 padded to 800
#define KT2  32           // K=1024
#define KT3  5            // K=138 padded to 160

union F8 { v8f v; float f[8]; };

// A-matrix 16-bit 16x32 fragment K mapping (ISA 7.12.2):
// lanes 0-15: V0=K{0,1}..V3=K{6,7}, V4-7=K{16..23}; lanes 16-31: +8 / +8
__device__ __forceinline__ int kitA(int hi, int e) {
  int j = e >> 1, t = e & 1;
  return (j < 4 ? 8 * hi + 2 * j : 16 + 8 * hi + 2 * (j - 4)) + t;
}
// B-matrix 32x16 fragment K mapping: same K order as A (lanes 0-15: K0-7,16-23; lanes 16-31: +8)
__device__ __forceinline__ int kitB(int hi, int e) { return kitA(hi, e); }

// Scatter one f32 value into packed A-fragment bf16 layout (for next GEMM).
__device__ __forceinline__ void pk_storeA(bf16* __restrict__ dst, int KtNext,
                                          int m, int n, float v) {
  int kt = n >> 5, q = n & 31;
  int hi, j2, t2;
  if (q < 16) { hi = q >> 3; j2 = (q & 7) >> 1; t2 = q & 1; }
  else        { int qq = q - 16; hi = qq >> 3; j2 = 4 + ((qq & 7) >> 1); t2 = qq & 1; }
  int e = 2 * j2 + t2;
  int lane2 = hi * 16 + (m & 15);
  dst[(((size_t)(m >> 4) * KtNext + kt) * 32 + lane2) * 16 + e] = (bf16)v;
}

// ---------------- pack kernels ----------------

// Patchify x + concat c, convert to bf16, packed A-fragment layout (K pad 800).
__global__ void pack_A1(const float* __restrict__ x, const float* __restrict__ c,
                        bf16* __restrict__ A1p) {
  int idx  = blockIdx.x * 256 + threadIdx.x;
  int e    = idx & 15;
  int lane = (idx >> 4) & 31;
  int tile = idx >> 9;
  int kt   = tile % KT1;
  int mt   = tile / KT1;
  int m    = mt * 16 + (lane & 15);
  int k    = kt * 32 + kitA(lane >> 4, e);
  int b = m >> 4, l = m & 15;
  float v = 0.f;
  if (k < PDIM) {
    int ch = k >> 8, kh = (k >> 4) & 15, kw = k & 15;
    int oh = l >> 2, ow = l & 3;
    v = x[((size_t)(b * 3 + ch) * 64 + oh * 16 + kh) * 64 + ow * 16 + kw];
  } else if (k < PDIM + NCLS) {
    v = c[b * NCLS + (k - PDIM)];
  }
  A1p[idx] = (bf16)v;
}

// Row-major W[Korig x N] f32 -> packed B-fragment bf16 tiles (zero-pad K).
__global__ void pack_B(const float* __restrict__ W, bf16* __restrict__ Bpk,
                       int Korig, int Kt, int N) {
  int idx  = blockIdx.x * 256 + threadIdx.x;
  int e    = idx & 15;
  int lane = (idx >> 4) & 31;
  int tile = idx >> 9;
  int kt   = tile % Kt;
  int nt   = tile / Kt;
  int n    = nt * 16 + (lane & 15);
  int k    = kt * 32 + kitB(lane >> 4, e);
  float v  = (k < Korig) ? W[(size_t)k * N + n] : 0.f;
  Bpk[idx] = (bf16)v;
}

// Fill decoder-input tail tile (k = 128..159): class condition + zero pad.
__global__ void fill_cb_A3(const float* __restrict__ c, bf16* __restrict__ A3p) {
  int idx  = blockIdx.x * 256 + threadIdx.x;   // MT * 512 elements
  int e    = idx & 15;
  int lane = (idx >> 4) & 31;
  int mt   = idx >> 9;                          // batch index == m tile
  int kin  = kitA(lane >> 4, e);
  float v  = (kin < NCLS) ? c[mt * NCLS + kin] : 0.f;
  A3p[((size_t)(mt * KT3 + 4) * 32 + lane) * 16 + e] = (bf16)v;
}

// ---------------- fused bf16 WMMA GEMM ----------------
// One wave computes a 64(M) x 64(N) block = 4x4 tiles of 16x16.
// EPI: 0 = bias+ReLU -> packed bf16 (encoder h)
//      1 = bias -> mu_logvar_2d f32 scatter + mu packed bf16 (decoder in)
//      2 = bias+ReLU -> packed bf16 (decoder hd)
//      3 = bias+sigmoid -> recon f32 unpatchify scatter
template <int EPI>
__global__ __launch_bounds__(32) void gemm_bf16(
    const bf16* __restrict__ Ap, const bf16* __restrict__ Bp,
    const float* __restrict__ bias, int Kt,
    bf16* __restrict__ outPk, int KtNext, float* __restrict__ outF) {
  const int lane = threadIdx.x;
  const int mt0  = blockIdx.x * 4;
  const int nt0  = blockIdx.y * 4;

  F8 acc[4][4];
#pragma unroll
  for (int i = 0; i < 4; ++i)
#pragma unroll
    for (int j = 0; j < 4; ++j)
#pragma unroll
      for (int r = 0; r < 8; ++r) acc[i][j].f[r] = 0.f;

  for (int kt = 0; kt < Kt; ++kt) {
    v16bf a[4], b[4];
#pragma unroll
    for (int i = 0; i < 4; ++i)
      a[i] = *(const v16bf*)(Ap + (((size_t)(mt0 + i) * Kt + kt) * 32 + lane) * 16);
#pragma unroll
    for (int j = 0; j < 4; ++j)
      b[j] = *(const v16bf*)(Bp + (((size_t)(nt0 + j) * Kt + kt) * 32 + lane) * 16);
#pragma unroll
    for (int i = 0; i < 4; ++i)
#pragma unroll
      for (int j = 0; j < 4; ++j)
        acc[i][j].v = __builtin_amdgcn_wmma_f32_16x16x32_bf16(
            false, a[i], false, b[j], (short)0, acc[i][j].v, false, false);
  }

  const int hi = lane >> 4;
#pragma unroll
  for (int j = 0; j < 4; ++j) {
    const int n  = (nt0 + j) * 16 + (lane & 15);
    const float bn = bias[n];
#pragma unroll
    for (int i = 0; i < 4; ++i) {
#pragma unroll
      for (int r = 0; r < 8; ++r) {
        int   m = (mt0 + i) * 16 + r + 8 * hi;
        float v = acc[i][j].f[r] + bn;
        if (EPI == 0 || EPI == 2) {
          v = v > 0.f ? v : 0.f;
          pk_storeA(outPk, KtNext, m, n, v);
        } else if (EPI == 1) {
          // mu_logvar_2d[b, n, oh, ow]: b = m>>4, l = m&15 (L == 16)
          outF[(size_t)(m >> 4) * 4096 + n * 16 + (m & 15)] = v;
          if (n < 128) pk_storeA(outPk, KtNext, m, n, v);  // mu -> decoder A
        } else {  // EPI == 3: sigmoid + unpatchify
          v = 1.f / (1.f + __expf(-v));
          int b = m >> 4, l = m & 15, oh = l >> 2, ow = l & 3;
          int ch = n >> 8, kh = (n >> 4) & 15, kw = n & 15;
          outF[((size_t)(b * 3 + ch) * 64 + oh * 16 + kh) * 64 + ow * 16 + kw] = v;
        }
      }
    }
  }
}

// ---------------- host launch ----------------
extern "C" void kernel_launch(void* const* d_in, const int* in_sizes, int n_in,
                              void* d_out, int out_size, void* d_ws, size_t ws_size,
                              hipStream_t stream) {
  const float* x      = (const float*)d_in[0];
  const float* c      = (const float*)d_in[1];
  const float* enc_w1 = (const float*)d_in[2];
  const float* enc_b1 = (const float*)d_in[3];
  const float* enc_w2 = (const float*)d_in[4];
  const float* enc_b2 = (const float*)d_in[5];
  const float* dec_w1 = (const float*)d_in[6];
  const float* dec_b1 = (const float*)d_in[7];
  const float* dec_w2 = (const float*)d_in[8];
  const float* dec_b2 = (const float*)d_in[9];

  char* ws = (char*)d_ws;
  bf16* A1p = (bf16*)(ws + 0);          // 26,214,400 B
  bf16* HDp = (bf16*)(ws + 0);          // 33,554,432 B, aliases A1p/B1p (dead by GEMM3)
  bf16* B1p = (bf16*)(ws + 26214400);   //  1,638,400 B
  bf16* Hp  = (bf16*)(ws + 27852800);   // 33,554,432 B
  bf16* B2p = (bf16*)(ws + 61407232);   //    524,288 B
  bf16* A3p = (bf16*)(ws + 61931520);   //  5,242,880 B
  bf16* B3p = (bf16*)(ws + 67174400);   //    327,680 B
  bf16* B4p = (bf16*)(ws + 67502080);   //  1,572,864 B  -> total 69,074,944 B

  float* recon = (float*)d_out;
  float* out2  = (float*)d_out + 12582912;  // mu_logvar_2d

  // operand packing (fp32 -> bf16, fragment-native tiled layout)
  pack_A1<<<51200, 256, 0, stream>>>(x, c, A1p);
  pack_B <<<3200,  256, 0, stream>>>(enc_w1, B1p, 778,  KT1, 1024);
  pack_B <<<1024,  256, 0, stream>>>(enc_w2, B2p, 1024, KT2, 256);
  pack_B <<<640,   256, 0, stream>>>(dec_w1, B3p, 138,  KT3, 1024);
  pack_B <<<3072,  256, 0, stream>>>(dec_w2, B4p, 1024, KT2, 768);
  fill_cb_A3<<<2048, 256, 0, stream>>>(c, A3p);

  // encoder: h = relu([p,c] W1 + b1) -> packed
  gemm_bf16<0><<<dim3(256, 16), 32, 0, stream>>>(A1p, B1p, enc_b1, KT1, Hp, KT2, nullptr);
  // mu_logvar = h W2 + b2 -> f32 output (transposed) + mu packed for decoder
  gemm_bf16<1><<<dim3(256, 4),  32, 0, stream>>>(Hp, B2p, enc_b2, KT2, A3p, KT3, out2);
  // decoder: hd = relu([mu,c] W1 + b1) -> packed
  gemm_bf16<2><<<dim3(256, 16), 32, 0, stream>>>(A3p, B3p, dec_b1, KT3, HDp, KT2, nullptr);
  // recon = sigmoid(hd W2 + b2), unpatchified
  gemm_bf16<3><<<dim3(256, 12), 32, 0, stream>>>(HDp, B4p, dec_b2, KT2, nullptr, 0, recon);
}
